// SparseMLP_Block_78700980732124
// MI455X (gfx1250) — matrix operand order repaired and checked by reading, not compiled, not run on hardware
//
#include <hip/hip_runtime.h>
#include <hip/hip_bf16.h>

#define TS 3136      // 56*56
#define NB 16
#define CC 256
#define C1 128

typedef __attribute__((ext_vector_type(16))) _Float16 v16h;
typedef __attribute__((ext_vector_type(8)))  _Float16 v8h;
typedef __attribute__((ext_vector_type(2)))  _Float16 v2h;
typedef __attribute__((ext_vector_type(8)))  float    v8f;
typedef __attribute__((ext_vector_type(4)))  float    v4f;
typedef __attribute__((ext_vector_type(2)))  float    v2f;

union HF16 { v16h v; v8h h[2]; };

__device__ __forceinline__ float gelu_exact(float x) {
  return 0.5f * x * (1.0f + erff(x * 0.70710678118654752f));
}

// CDNA5 async global->LDS copy (ASYNCcnt-tracked), per CDNA5 ISA
__device__ __forceinline__ void async_copy_f32(const float* gsrc, float* ldst) {
  unsigned loff = (unsigned)(size_t)ldst;   // generic LDS ptr low 32 bits = LDS offset
  asm volatile("global_load_async_to_lds_b32 %0, %1, off"
               :: "v"(loff), "v"(gsrc) : "memory");
}
__device__ __forceinline__ void async_wait0() {
  asm volatile("s_wait_asynccnt 0x0" ::: "memory");
}

// ---------------------------------------------------------------------------
// Fused pointwise GEMM over NCHW:  out[n,oc,s] = epi( sum_k A[n,k,s] * W[oc,k] )
// A is a 2-segment channel-concat view: segment = ptr + n*batchStride + k*TS + s
// K=256. BM=128, BN=128, BK=32, 256 threads (8 waves, 2x4), LDS ping-pong.
// ksplit is a multiple of 32 -> each BK tile lives wholly in one segment.
// ---------------------------------------------------------------------------
__global__ __launch_bounds__(256)
void gemm_pw_kernel(const float* __restrict__ a0, long a0bs,
                    const float* __restrict__ a1, long a1bs, int ksplit,
                    const float* __restrict__ Wt,
                    const float* __restrict__ bias,
                    const float* __restrict__ bng, const float* __restrict__ bnb,
                    const float* __restrict__ bnm, const float* __restrict__ bnv,
                    int doGelu,
                    const float* __restrict__ pos, int posLimit,
                    float* __restrict__ outp, long outbs)
{
  __shared__ __align__(16) float ldsC[128 * 132];     // 67.5KB; also holds ping-pong A/B
  _Float16* ldsH = (_Float16*)ldsC;   // buffer b: A at b*8192, B at b*8192+4096 (halves)

  const int tid = threadIdx.x;
  const int m0 = blockIdx.x * 128;
  const int n0 = blockIdx.y * 128;
  const int lane = tid & 31;
  const int wv   = tid >> 5;
  const int wvN  = wv & 3;    // 32-wide N slice
  const int wvM  = wv >> 2;   // 64-wide M slice

  v8f acc[4][2];
#pragma unroll
  for (int mi = 0; mi < 4; ++mi)
#pragma unroll
    for (int ni = 0; ni < 2; ++ni)
#pragma unroll
      for (int i = 0; i < 8; ++i) acc[mi][ni][i] = 0.0f;

  // staging coordinates: A pairs (k,k+1) per thread; B 8-byte global loads
  const int mIn    = tid & 127;       // A row
  const int kHalf  = tid >> 7;        // A k-block (0: k 0-15, 1: k 16-31)
  const int nInB   = tid >> 4;        // B o base (0..15)
  const int kPairB = (tid & 15) * 2;  // B k pair

  const int mg   = m0 + mIn;
  const int nbat = mg / TS;
  const int ssp  = mg - nbat * TS;
  const long baseA0 = (long)nbat * a0bs + ssp;
  const long baseA1 = (long)nbat * a1bs + ssp;

  auto stage = [&](int kt, int b) {
    const int k0 = kt * 32;
    _Float16* A = ldsH + b * 8192;
    _Float16* B = A + 4096;
    // uniform per-tile segment select: one 64-bit base, immediate offsets after
    const bool  s0  = (k0 < ksplit);
    const float* pA = s0 ? a0 : a1;
    const long  base = s0 ? baseA0 : baseA1;
    const int   kb   = (s0 ? k0 : (k0 - ksplit)) + kHalf * 16;
    const float* src = pA + base + (long)kb * TS;
    _Float16*   dstA = A + mIn * 32 + kHalf * 16;
#pragma unroll
    for (int j = 0; j < 8; ++j) {           // 2 coalesced b32 loads -> 1 b32 ds store
      float va = src[(long)(2 * j) * TS];
      float vb = src[(long)(2 * j + 1) * TS];
      v2h pk; pk.x = (_Float16)va; pk.y = (_Float16)vb;
      *(v2h*)(dstA + j * 2) = pk;
    }
    const float* wsrc = Wt + (long)(n0 + nInB) * 256 + (k0 + kPairB);
#pragma unroll
    for (int j = 0; j < 8; ++j) {           // 1 b64 load -> 1 b32 ds store
      v2f wv2 = *(const v2f*)(wsrc + (long)j * (16 * 256));
      v2h pk; pk.x = (_Float16)wv2.x; pk.y = (_Float16)wv2.y;
      *(v2h*)(B + (nInB + j * 16) * 32 + kPairB) = pk;
    }
  };

  stage(0, 0);

  const int rlo   = lane & 15;
  const int koffA = (lane < 16) ? 0 : 8;    // A frag: K 0-7/16-23 vs 8-15/24-31
  const int koffB = (lane < 16) ? 0 : 8;    // B frag: K 0-7/16-23 vs 8-15/24-31

  for (int kt = 0; kt < 8; ++kt) {
    __syncthreads();
    const _Float16* A = ldsH + (kt & 1) * 8192;
    const _Float16* B = A + 4096;
    HF16 afr[4], bfr[2];
#pragma unroll
    for (int mi = 0; mi < 4; ++mi) {
      const _Float16* pa = A + (wvM * 64 + mi * 16 + rlo) * 32;
      afr[mi].h[0] = *(const v8h*)(pa + koffA);
      afr[mi].h[1] = *(const v8h*)(pa + koffA + 16);
    }
#pragma unroll
    for (int ni = 0; ni < 2; ++ni) {
      const _Float16* pb = B + (wvN * 32 + ni * 16 + rlo) * 32 + koffB;
      bfr[ni].h[0] = *(const v8h*)(pb);
      bfr[ni].h[1] = *(const v8h*)(pb + 16);
    }
#pragma unroll
    for (int mi = 0; mi < 4; ++mi)
#pragma unroll
      for (int ni = 0; ni < 2; ++ni)
        acc[mi][ni] = __builtin_amdgcn_wmma_f32_16x16x32_f16(
            false, afr[mi].v, false, bfr[ni].v, (short)0, acc[mi][ni], false, false);
    if (kt < 7) stage(kt + 1, (kt + 1) & 1);   // prefetch next tile under the WMMAs
  }

  __syncthreads();
  // bounce C through LDS [o][m] (pitch 132: 16B-aligned rows, 4-bank row skew)
#pragma unroll
  for (int mi = 0; mi < 4; ++mi)
#pragma unroll
    for (int ni = 0; ni < 2; ++ni) {
      int mbase = wvM * 64 + mi * 16 + ((lane < 16) ? 0 : 8);
      int o     = wvN * 32 + ni * 16 + (lane & 15);
#pragma unroll
      for (int i = 0; i < 8; ++i) ldsC[o * 132 + mbase + i] = acc[mi][ni][i];
    }
  __syncthreads();

  // epilogue + coalesced float4 NCHW stores
  for (int rep = 0; rep < 16; ++rep) {
    int idx = rep * 256 + tid;          // 4096 float4 groups
    int o   = idx >> 5;
    int mq  = (idx & 31) * 4;
    int oc  = n0 + o;
    v4f v = *(const v4f*)&ldsC[o * 132 + mq];
    int mg2 = m0 + mq;
    int n2  = mg2 / TS;                 // group never straddles batch (TS%4==0)
    int s2  = mg2 - n2 * TS;
    if (bng) {
      float sc = bng[oc] * rsqrtf(bnv[oc] + 1e-5f);
      float sh = bnb[oc] - bnm[oc] * sc;
      if (bias) sh += bias[oc] * sc;
#pragma unroll
      for (int i = 0; i < 4; ++i) v[i] = v[i] * sc + sh;
    }
    if (doGelu) {
#pragma unroll
      for (int i = 0; i < 4; ++i) v[i] = gelu_exact(v[i]);
    }
    if (pos && oc < posLimit) {
#pragma unroll
      for (int i = 0; i < 4; ++i) {
        int s3 = s2 + i;
        int h2 = s3 / 56, w2 = s3 - h2 * 56;
        v[i] += pos[(w2 - h2 + 56) * C1 + oc];   // rel-pos idx = w - h + 56
      }
    }
    *(v4f*)&outp[(long)n2 * outbs + (long)oc * TS + s2] = v;
  }
}

// ---------------------------------------------------------------------------
// Strip convolutions. mode 0: H-mix on x1[:, :64]; mode 1: W-mix on x1[:, 64:128]
// Plane + weights staged with async global->LDS.
// ---------------------------------------------------------------------------
__global__ __launch_bounds__(256)
void strip_kernel(const float* __restrict__ Y,
                  const float* __restrict__ wh, const float* __restrict__ bh,
                  const float* __restrict__ wwp, const float* __restrict__ bw,
                  float* __restrict__ outBuf)
{
  const int g = blockIdx.x, n = blockIdx.y, mode = blockIdx.z;
  const int tid = threadIdx.x;
  __shared__ float inp[58 * 58];       // padded +1 both dims
  __shared__ float wgt[56 * 168];      // [oo][ii*3 + t]
  const float* wsel = mode ? wwp : wh;
  const float* bsel = mode ? bw : bh;
  const float* src  = Y + ((long)(n * CC + mode * 64 + g)) * TS;
  float*       dst  = outBuf + ((long)(n * C1 + mode * 64 + g)) * TS;

  for (int i = tid; i < 58 * 58; i += 256) inp[i] = 0.f;
  __syncthreads();
  for (int i = tid; i < TS; i += 256) {
    int h = i / 56, w = i - h * 56;
    async_copy_f32(src + i, &inp[(h + 1) * 58 + (w + 1)]);
  }
  for (int i = tid; i < 56 * 168; i += 256)
    async_copy_f32(wsel + (long)g * 9408 + i, &wgt[i]);
  async_wait0();
  __syncthreads();

  for (int i = tid; i < TS; i += 256) {
    int o1 = i / 56, o2 = i - o1 * 56;
    float acc2 = bsel[g * 56 + (mode ? o2 : o1)];
    if (mode == 0) {                      // out[ho=o1, w=o2]: sum over hi, taps along w
      const float* wr = wgt + o1 * 168;
#pragma unroll 4
      for (int ii = 0; ii < 56; ++ii) {
        const float* ir = inp + (ii + 1) * 58 + o2;
        acc2 += ir[0] * wr[ii * 3] + ir[1] * wr[ii * 3 + 1] + ir[2] * wr[ii * 3 + 2];
      }
    } else {                              // out[h=o1, wo=o2]: sum over wi, taps along h
      const float* wr = wgt + o2 * 168;
#pragma unroll 4
      for (int ii = 0; ii < 56; ++ii) {
        const float* ic = inp + o1 * 58 + (ii + 1);
        acc2 += ic[0] * wr[ii * 3] + ic[58] * wr[ii * 3 + 1] + ic[116] * wr[ii * 3 + 2];
      }
    }
    dst[i] = acc2;
  }
}

// ---------------------------------------------------------------------------
// Depthwise 3x7 + 7x3 convs on x2 + per-plane mean -> att (async plane staging)
// ---------------------------------------------------------------------------
__global__ __launch_bounds__(256)
void dw_att_kernel(const float* __restrict__ Y,
                   const float* __restrict__ fch, const float* __restrict__ fcw,
                   float* __restrict__ H2, float* __restrict__ W2,
                   float* __restrict__ att)
{
  const int c = blockIdx.x, n = blockIdx.y, tid = threadIdx.x;
  __shared__ float inp[62 * 62];   // pad 3
  __shared__ float kh[21], kw[21];
  __shared__ float red[256];
  const float* src = Y + ((long)(n * CC + C1 + c)) * TS;
  float* dh  = H2 + ((long)(n * C1 + c)) * TS;
  float* dwp = W2 + ((long)(n * C1 + c)) * TS;

  for (int i = tid; i < 62 * 62; i += 256) inp[i] = 0.f;
  __syncthreads();
  for (int i = tid; i < TS; i += 256) {
    int h = i / 56, w = i - h * 56;
    async_copy_f32(src + i, &inp[(h + 3) * 62 + (w + 3)]);
  }
  if (tid < 21) { kh[tid] = fch[c * 21 + tid]; kw[tid] = fcw[c * 21 + tid]; }
  async_wait0();
  __syncthreads();

  float lsum = 0.f;
  for (int i = tid; i < TS; i += 256) {
    int h = i / 56, w = i - h * 56;
    const float* base = inp + (h + 3) * 62 + (w + 3);
    float a = 0.f, b = 0.f;
#pragma unroll
    for (int d1 = -1; d1 <= 1; ++d1)
#pragma unroll
      for (int d2 = -3; d2 <= 3; ++d2)
        a += base[d1 * 62 + d2] * kh[(d1 + 1) * 7 + (d2 + 3)];
#pragma unroll
    for (int d1 = -3; d1 <= 3; ++d1)
#pragma unroll
      for (int d2 = -1; d2 <= 1; ++d2)
        b += base[d1 * 62 + d2] * kw[(d1 + 3) * 3 + (d2 + 1)];
    dh[i] = a; dwp[i] = b;
    lsum += a + b + base[0];
  }
  red[tid] = lsum;
  __syncthreads();
  for (int st = 128; st > 0; st >>= 1) {
    if (tid < st) red[tid] += red[tid + st];
    __syncthreads();
  }
  if (tid == 0) att[n * C1 + c] = red[0] * (1.0f / TS);
}

// ---------------------------------------------------------------------------
// Tiny reweight MLP + softmax over triples -> A3[n, c, 3]
// ---------------------------------------------------------------------------
__global__ __launch_bounds__(256)
void attn_mix_kernel(const float* __restrict__ att,
                     const float* __restrict__ w1, const float* __restrict__ b1,
                     const float* __restrict__ w2, const float* __restrict__ b2,
                     float* __restrict__ A3)
{
  __shared__ float h1[16 * 64];
  const int tid = threadIdx.x;
  for (int idx = tid; idx < 16 * 64; idx += 256) {
    int n = idx >> 6, j = idx & 63;
    float acc = b1[j];
    for (int k = 0; k < 128; ++k) acc += att[n * 128 + k] * w1[j * 128 + k];
    h1[idx] = gelu_exact(acc);
  }
  __syncthreads();
  for (int idx = tid; idx < 16 * 128; idx += 256) {
    int n = idx >> 7, cc = idx & 127;
    float e[3];
#pragma unroll
    for (int r = 0; r < 3; ++r) {
      int row = cc * 3 + r;
      float acc = b2[row];
      for (int k = 0; k < 64; ++k) acc += h1[n * 64 + k] * w2[row * 64 + k];
      e[r] = acc;
    }
    float mx = fmaxf(e[0], fmaxf(e[1], e[2]));
    float e0 = expf(e[0] - mx), e1 = expf(e[1] - mx), e2 = expf(e[2] - mx);
    float inv = 1.f / (e0 + e1 + e2);
    A3[idx * 3 + 0] = e0 * inv; A3[idx * 3 + 1] = e1 * inv; A3[idx * 3 + 2] = e2 * inv;
  }
}

// ---------------------------------------------------------------------------
// x2_final = x2h*a0 + x2w*a1 + x2*a2
// ---------------------------------------------------------------------------
__global__ __launch_bounds__(256)
void x2_mix_kernel(const float* __restrict__ Y, const float* __restrict__ H2,
                   const float* __restrict__ W2, const float* __restrict__ A3,
                   float* __restrict__ X2F)
{
  long e = (long)blockIdx.x * 256 + threadIdx.x;       // < 16*128*3136
  int nc = (int)(e / TS);
  int s  = (int)(e - (long)nc * TS);
  int n  = nc >> 7, c = nc & 127;
  const float* a = A3 + nc * 3;
  float x2 = Y[((long)(n * CC + C1 + c)) * TS + s];
  X2F[e] = H2[e] * a[0] + W2[e] * a[1] + x2 * a[2];
}

// ---------------------------------------------------------------------------
extern "C" void kernel_launch(void* const* d_in, const int* in_sizes, int n_in,
                              void* d_out, int out_size, void* d_ws, size_t ws_size,
                              hipStream_t stream)
{
  const float* x        = (const float*)d_in[0];
  const float* mlp_w    = (const float*)d_in[1];
  const float* mlp_b    = (const float*)d_in[2];
  const float* mlp_bn_g = (const float*)d_in[3];
  const float* mlp_bn_b = (const float*)d_in[4];
  const float* mlp_bn_m = (const float*)d_in[5];
  const float* mlp_bn_v = (const float*)d_in[6];
  const float* pos_h_t  = (const float*)d_in[7];
  const float* pos_w_t  = (const float*)d_in[8];
  const float* proj_h_w = (const float*)d_in[9];
  const float* proj_h_b = (const float*)d_in[10];
  const float* proj_w_w = (const float*)d_in[11];
  const float* proj_w_b = (const float*)d_in[12];
  const float* fuse_h_w = (const float*)d_in[13];
  const float* bn_g     = (const float*)d_in[14];
  const float* bn_b     = (const float*)d_in[15];
  const float* bn_m     = (const float*)d_in[16];
  const float* bn_v     = (const float*)d_in[17];
  const float* fuse_w_w = (const float*)d_in[18];
  const float* fc_h_w   = (const float*)d_in[19];
  const float* fc_w_w   = (const float*)d_in[20];
  const float* rw_w1    = (const float*)d_in[21];
  const float* rw_b1    = (const float*)d_in[22];
  const float* rw_w2    = (const float*)d_in[23];
  const float* rw_b2    = (const float*)d_in[24];
  const float* fuse_out_w = (const float*)d_in[25];

  float* out = (float*)d_out;
  float* Y = out;    // post-MLP activations live in d_out; dead before final GEMM

  const long SEG = (long)NB * C1 * TS;  // 6,422,528 floats
  float* B0 = (float*)d_ws;             // STRIP, later X1C
  float* B1 = B0 + SEG;                 // X1B, later X2F
  float* B2 = B1 + SEG;                 // x2h
  float* B3 = B2 + SEG;                 // x2w
  float* att = B3 + SEG;                // 2048
  float* A3  = att + NB * C1;           // 6144

  const long bsX = (long)CC * TS;
  const long bsH = (long)C1 * TS;
  dim3 blk(256);

  // 1) y = gelu(bn(x @ mlp_w^T + mlp_b)); y[:, :128] += pos_h   -> Y
  gemm_pw_kernel<<<dim3(392, 2), blk, 0, stream>>>(
      x, bsX, x, bsX, 256, mlp_w,
      mlp_b, mlp_bn_g, mlp_bn_b, mlp_bn_m, mlp_bn_v,
      1, pos_h_t, 128, Y, bsX);

  // 2) strip convs: xh (ch 0-63), xw (ch 64-127) -> B0
  strip_kernel<<<dim3(64, 16, 2), blk, 0, stream>>>(
      Y, proj_h_w, proj_h_b, proj_w_w, proj_w_b, B0);

  // 3) x1 = gelu(bn(concat(xh,xw,x1) @ fuse_h_w^T)) + pos_w -> B1
  gemm_pw_kernel<<<dim3(392, 1), blk, 0, stream>>>(
      B0, bsH, Y, bsX, 128, fuse_h_w,
      nullptr, bn_g, bn_b, bn_m, bn_v,
      1, pos_w_t, 128, B1, bsH);

  // 4) x1 = concat(x1, x2) @ fuse_w_w^T -> B0 (X1C)
  gemm_pw_kernel<<<dim3(392, 1), blk, 0, stream>>>(
      B1, bsH, Y + (long)C1 * TS, bsX, 128, fuse_w_w,
      nullptr, nullptr, nullptr, nullptr, nullptr,
      0, nullptr, 0, B0, bsH);

  // 5) depthwise convs + attention mean -> B2, B3, att
  dw_att_kernel<<<dim3(128, 16), blk, 0, stream>>>(Y, fc_h_w, fc_w_w, B2, B3, att);

  // 6) reweight MLP + softmax -> A3
  attn_mix_kernel<<<dim3(1), blk, 0, stream>>>(att, rw_w1, rw_b1, rw_w2, rw_b2, A3);

  // 7) x2 = x2h*a0 + x2w*a1 + x2*a2 -> B1 (X2F)
  x2_mix_kernel<<<dim3(25088), blk, 0, stream>>>(Y, B2, B3, A3, B1);

  // 8) out = concat(x1, x2) @ fuse_out_w^T -> d_out (Y is dead)
  gemm_pw_kernel<<<dim3(392, 2), blk, 0, stream>>>(
      B0, bsH, B1, bsH, 128, fuse_out_w,
      nullptr, nullptr, nullptr, nullptr, nullptr,
      0, nullptr, 0, out, bsX);
}
